// RNNTransition_24318104830131
// MI455X (gfx1250) — matrix operand order repaired and checked by reading, not compiled, not run on hardware
//
#include <hip/hip_runtime.h>
#include <hip/hip_bf16.h>

// ---------------------------------------------------------------------------
// Problem constants (from reference): T=32, B=256, C=64, S=7, H=1024, depth=3
// ---------------------------------------------------------------------------
#define TT   32
#define BB   256
#define HH   1024
#define K1   3136            // C*S*S
#define MBIG 8192            // T*B

typedef __attribute__((ext_vector_type(16))) __bf16 v16bf;
typedef __attribute__((ext_vector_type(8)))  __bf16 v8bf;
typedef __attribute__((ext_vector_type(4)))  __bf16 v4bf;
typedef __attribute__((ext_vector_type(8)))  float  v8f;

// ---------------------------------------------------------------------------
// fp32 -> bf16 conversion (weights)
// ---------------------------------------------------------------------------
__global__ void cvt_f32_bf16(const float* __restrict__ src, __bf16* __restrict__ dst,
                             size_t n) {
    size_t i = (size_t)blockIdx.x * blockDim.x + threadIdx.x;
    size_t stride = (size_t)gridDim.x * blockDim.x;
    for (; i < n; i += stride) dst[i] = (__bf16)src[i];
}

// ---------------------------------------------------------------------------
// Tiled WMMA GEMM:  D[M,N] = act(A[M,K] @ B[N,K]^T + bias[N])
//   A: row-major [M,K], TA = float (converted to bf16 during staging) or __bf16
//   B: row-major [N,K], always bf16 (weights pre-converted)
//   D: row-major [M,ldd], TD = float or __bf16
// Block tile 128x128, BK=32, 256 threads = 8 waves, wave tile 32x64 (2x4 WMMA).
// LDS double-buffered; next K-tile is prefetched into registers while the
// current tile's WMMAs run, then committed to the other LDS buffer -> one
// barrier per K-iteration and global latency overlapped with matrix math.
// ---------------------------------------------------------------------------
#define BM 128
#define BN 128
#define BK 32
#define LPA 40   // LDS pitch (bf16 elems) for A tile rows (32 + 8 pad)
#define LPB 40   // LDS pitch for B tile rows

template<typename TA, bool RELU, typename TD>
__global__ __launch_bounds__(256)
void gemm_wmma_bf16(const TA* __restrict__ A, const __bf16* __restrict__ Bw,
                    const float* __restrict__ bias, TD* __restrict__ D,
                    int M, int N, int K, int lda, int ldb, int ldd) {
    __shared__ __align__(16) __bf16 As[2][BM * LPA];
    __shared__ __align__(16) __bf16 Bs[2][BN * LPB];

    const int tid  = threadIdx.x;
    const int lane = tid & 31;
    const int wave = tid >> 5;       // 0..7
    const int wm   = wave & 3;       // wave row   (4 * 32 = 128)
    const int wn   = wave >> 2;      // wave col   (2 * 64 = 128)
    const int lr   = lane & 15;      // lane row/col inside 16
    const int kh   = lane >> 4;      // K-half selector (ISA 16-bit A/B layout)

    const int blockM = blockIdx.y * BM;
    const int blockN = blockIdx.x * BN;

    v8f acc[2][4];
    v8f zero = {};
#pragma unroll
    for (int i = 0; i < 2; ++i)
#pragma unroll
        for (int j = 0; j < 4; ++j) acc[i][j] = zero;

    // register staging for the prefetched K-tile
    float4 arf[4];    // A tile when TA == float  (128x32 f32 = 4 float4/thread)
    v8bf   arb[2];    // A tile when TA == __bf16 (128x32 bf16 = 2 v8bf/thread)
    v8bf   brg[2];    // B tile (always bf16)

    auto g_load = [&](int k0) {
        if constexpr (sizeof(TA) == 4) {
#pragma unroll
            for (int j = 0; j < 4; ++j) {
                int c = tid + j * 256;             // 0..1023 float4-chunks
                arf[j] = *(const float4*)(A + (size_t)(blockM + (c >> 3)) * lda
                                          + k0 + (c & 7) * 4);
            }
        } else {
#pragma unroll
            for (int j = 0; j < 2; ++j) {
                int c = tid + j * 256;             // 0..511 v8bf-chunks
                arb[j] = *(const v8bf*)((const __bf16*)A
                                        + (size_t)(blockM + (c >> 2)) * lda
                                        + k0 + (c & 3) * 8);
            }
        }
#pragma unroll
        for (int j = 0; j < 2; ++j) {
            int c = tid + j * 256;
            brg[j] = *(const v8bf*)(Bw + (size_t)(blockN + (c >> 2)) * ldb
                                    + k0 + (c & 3) * 8);
        }
    };

    auto l_store = [&](int buf) {
        if constexpr (sizeof(TA) == 4) {
#pragma unroll
            for (int j = 0; j < 4; ++j) {
                int c = tid + j * 256;
                v4bf o = { (__bf16)arf[j].x, (__bf16)arf[j].y,
                           (__bf16)arf[j].z, (__bf16)arf[j].w };
                *(v4bf*)&As[buf][(c >> 3) * LPA + (c & 7) * 4] = o;
            }
        } else {
#pragma unroll
            for (int j = 0; j < 2; ++j) {
                int c = tid + j * 256;
                *(v8bf*)&As[buf][(c >> 2) * LPA + (c & 3) * 8] = arb[j];
            }
        }
#pragma unroll
        for (int j = 0; j < 2; ++j) {
            int c = tid + j * 256;
            *(v8bf*)&Bs[buf][(c >> 2) * LPB + (c & 3) * 8] = brg[j];
        }
    };

    // prologue: stage tile 0
    g_load(0);
    l_store(0);
    __syncthreads();

    int cur = 0;
    for (int k0 = 0; k0 < K; k0 += BK) {
        const bool has_next = (k0 + BK) < K;
        if (has_next) g_load(k0 + BK);             // prefetch next tile (global)

        // ---- fragments from LDS[cur] (ISA 7.12.2 bf16 A 16x32 / B 32x16) ----
        v16bf af[2];
#pragma unroll
        for (int i = 0; i < 2; ++i) {
            const __bf16* pa = &As[cur][(wm * 32 + i * 16 + lr) * LPA + kh * 8];
            v8bf lo = *(const v8bf*)pa;            // K = kh*8 .. +7
            v8bf hi = *(const v8bf*)(pa + 16);     // K = kh*8+16 .. +7
            af[i] = __builtin_shufflevector(lo, hi, 0,1,2,3,4,5,6,7,
                                                    8,9,10,11,12,13,14,15);
        }
        v16bf bfr[4];
#pragma unroll
        for (int j = 0; j < 4; ++j) {
            const __bf16* pb = &Bs[cur][(wn * 64 + j * 16 + lr) * LPB + kh * 8];
            v8bf b0 = *(const v8bf*)pb;            // K = kh*8 .. +7
            v8bf b1 = *(const v8bf*)(pb + 16);     // K = kh*8+16 .. +7
            bfr[j] = __builtin_shufflevector(b0, b1, 0,1,2,3,4,5,6,7,
                                                     8,9,10,11,12,13,14,15);
        }

        // ---- 2x4 WMMAs ----
#pragma unroll
        for (int i = 0; i < 2; ++i)
#pragma unroll
            for (int j = 0; j < 4; ++j)
                acc[i][j] = __builtin_amdgcn_wmma_f32_16x16x32_bf16(
                    false, af[i], false, bfr[j], (short)0, acc[i][j], false, false);

        if (has_next) {
            l_store(cur ^ 1);                      // commit prefetched tile
            __syncthreads();                       // single barrier / iteration
            cur ^= 1;
        }
    }

    // ---- epilogue: bias (+ReLU), store per 16x16 f32 C/D lane layout ----
#pragma unroll
    for (int i = 0; i < 2; ++i) {
#pragma unroll
        for (int j = 0; j < 4; ++j) {
            int col  = blockN + wn * 64 + j * 16 + lr;
            int row0 = blockM + wm * 32 + i * 16 + kh * 8;
            float bv = bias[col];
#pragma unroll
            for (int r = 0; r < 8; ++r) {
                float v = acc[i][j][r] + bv;
                if (RELU) v = fmaxf(v, 0.0f);
                D[(size_t)(row0 + r) * ldd + col] = (TD)v;
            }
        }
    }
}

// ---------------------------------------------------------------------------
// h *= mask[t, b]   (episode reset)
// ---------------------------------------------------------------------------
__global__ void mask_mul(float* __restrict__ h, const float* __restrict__ mt, int n) {
    int i = blockIdx.x * blockDim.x + threadIdx.x;
    if (i < n) h[i] *= mt[i >> 10];                 // H == 1024
}

// ---------------------------------------------------------------------------
// Fused GRU gate update:
//   r = sig(gi_r+gh_r); z = sig(gi_z+gh_z); n = tanh(gi_n + r*gh_n)
//   h' = (1-z)*n + z*h     (optionally also written to outs[t])
// ---------------------------------------------------------------------------
__global__ void gru_update(const float* __restrict__ gi, const float* __restrict__ gh,
                           const float* __restrict__ h_in, float* __restrict__ h_out,
                           float* __restrict__ outs_opt, int n) {
    int i = blockIdx.x * blockDim.x + threadIdx.x;
    if (i >= n) return;
    int b = i >> 10, j = i & 1023;
    size_t g = (size_t)b * (3 * HH) + j;
    float r  = gi[g]            + gh[g];
    float z  = gi[g + HH]       + gh[g + HH];
    float gn = gi[g + 2 * HH];
    float hn = gh[g + 2 * HH];
    r = 1.0f / (1.0f + __expf(-r));
    z = 1.0f / (1.0f + __expf(-z));
    float nn = tanhf(gn + r * hn);
    float ho = (1.0f - z) * nn + z * h_in[i];
    h_out[i] = ho;
    if (outs_opt) outs_opt[i] = ho;
}

// ---------------------------------------------------------------------------
// Host orchestration
// ---------------------------------------------------------------------------
extern "C" void kernel_launch(void* const* d_in, const int* in_sizes, int n_in,
                              void* d_out, int out_size, void* d_ws, size_t ws_size,
                              hipStream_t stream) {
    (void)in_sizes; (void)n_in; (void)out_size;
    const float* x      = (const float*)d_in[0];   // [8192, 3136]
    const float* hxs    = (const float*)d_in[1];   // [256, 1024]
    const float* masks  = (const float*)d_in[2];   // [8192]
    const float* w_lin  = (const float*)d_in[3];   // [1024, 3136]
    const float* b_lin  = (const float*)d_in[4];   // [1024]
    const float* w_ih   = (const float*)d_in[5];   // [3072, 1024]
    const float* w_hh   = (const float*)d_in[6];   // [3072, 1024]
    const float* b_ih   = (const float*)d_in[7];   // [3072]
    const float* b_hh   = (const float*)d_in[8];   // [3072]

    float* outs    = (float*)d_out;                        // [8192, 1024]
    float* h_final = (float*)d_out + (size_t)MBIG * HH;    // [256, 1024]

    // ---- workspace carve ----
    char* p = (char*)d_ws;
    auto carve = [&](size_t bytes) -> char* {
        char* r = p; p += (bytes + 255) & ~(size_t)255; return r;
    };
    __bf16* wlin_bf = (__bf16*)carve((size_t)HH * K1 * 2);
    __bf16* wih_bf  = (__bf16*)carve((size_t)3 * HH * HH * 2);
    __bf16* whh_bf  = (__bf16*)carve((size_t)3 * HH * HH * 2);
    __bf16* feat_bf = (__bf16*)carve((size_t)MBIG * HH * 2);
    float*  h       = (float*) carve((size_t)BB * HH * 4);
    float*  gh      = (float*) carve((size_t)BB * 3 * HH * 4);
    size_t base_used = (size_t)(p - (char*)d_ws);
    size_t gi_all_b  = (size_t)MBIG * 3 * HH * 4;          // ~100 MB
    bool precompute  = (base_used + gi_all_b) <= ws_size;
    float* gi = (float*)carve(precompute ? gi_all_b : (size_t)BB * 3 * HH * 4);

    // ---- convert weights to bf16 ----
    cvt_f32_bf16<<<512, 256, 0, stream>>>(w_lin, wlin_bf, (size_t)HH * K1);
    cvt_f32_bf16<<<512, 256, 0, stream>>>(w_ih,  wih_bf,  (size_t)3 * HH * HH);
    cvt_f32_bf16<<<512, 256, 0, stream>>>(w_hh,  whh_bf,  (size_t)3 * HH * HH);

    // ---- GEMM 1: feat = relu(x @ w_lin.T + b_lin) -> bf16 ----
    gemm_wmma_bf16<float, true, __bf16>
        <<<dim3(HH / BN, MBIG / BM), 256, 0, stream>>>(
            x, wlin_bf, b_lin, feat_bf, MBIG, HH, K1, K1, K1, HH);

    // ---- GEMM 2 (hoisted out of the scan): gi = feat @ w_ih.T + b_ih ----
    if (precompute) {
        gemm_wmma_bf16<__bf16, false, float>
            <<<dim3(3 * HH / BN, MBIG / BM), 256, 0, stream>>>(
                feat_bf, wih_bf, b_ih, gi, MBIG, 3 * HH, HH, HH, HH, 3 * HH);
    }

    // ---- h = hxs ----
    hipMemcpyAsync(h, hxs, (size_t)BB * HH * 4, hipMemcpyDeviceToDevice, stream);

    const int nBH = BB * HH;
    for (int t = 0; t < TT; ++t) {
        const float* gi_t = precompute ? gi + (size_t)t * BB * 3 * HH : gi;
        if (!precompute) {
            gemm_wmma_bf16<__bf16, false, float>
                <<<dim3(3 * HH / BN, BB / BM), 256, 0, stream>>>(
                    feat_bf + (size_t)t * BB * HH, wih_bf, b_ih, gi,
                    BB, 3 * HH, HH, HH, HH, 3 * HH);
        }
        mask_mul<<<nBH / 256, 256, 0, stream>>>(h, masks + (size_t)t * BB, nBH);
        for (int it = 0; it < 3; ++it) {
            // gh = h @ w_hh.T + b_hh   (w_hh is L2-resident: 6.3 MB bf16)
            gemm_wmma_bf16<float, false, float>
                <<<dim3(3 * HH / BN, BB / BM), 256, 0, stream>>>(
                    h, whh_bf, b_hh, gh, BB, 3 * HH, HH, HH, HH, 3 * HH);
            gru_update<<<nBH / 256, 256, 0, stream>>>(
                gi_t, gh, h, h, (it == 2) ? outs + (size_t)t * BB * HH : nullptr, nBH);
        }
    }

    hipMemcpyAsync(h_final, h, (size_t)BB * HH * 4, hipMemcpyDeviceToDevice, stream);
}
